// Aggregate_75239237092020
// MI455X (gfx1250) — matrix operand order repaired and checked by reading, not compiled, not run on hardware
//
#include <hip/hip_runtime.h>
#include <hip/hip_bf16.h>

#define HW    3072   // 48*64
#define HH    48
#define WW    64
#define NBN   8      // b(2) * heads(4)

typedef __attribute__((ext_vector_type(16))) __bf16         v16bf;
typedef __attribute__((ext_vector_type(16))) unsigned short v16u;
typedef __attribute__((ext_vector_type(8)))  float          v8f;
typedef __attribute__((ext_vector_type(4)))  unsigned int   u32x4;
typedef __attribute__((ext_vector_type(8)))  int            i32x8;
typedef __attribute__((ext_vector_type(4)))  int            i32x4;

#if defined(__has_builtin)
#if __has_builtin(__builtin_amdgcn_tensor_load_to_lds) && \
    __has_builtin(__builtin_amdgcn_s_wait_tensorcnt)
#define HAVE_TDM 1
#endif
#endif
#ifndef HAVE_TDM
#define HAVE_TDM 0
#endif
#if HAVE_TDM && __has_include(<hip/amd_detail/amd_gfx1250_TDM.h>)
#define TDM_6ARG 1
#else
#define TDM_6ARG 0
#endif

static __device__ __forceinline__ unsigned short f2bf(float f) {
    unsigned int u = __builtin_bit_cast(unsigned int, f);
    unsigned int r = u + 0x7FFFu + ((u >> 16) & 1u);
    return (unsigned short)(r >> 16);
}
static __device__ __forceinline__ float bf2f(unsigned short s) {
    return __builtin_bit_cast(float, (unsigned int)s << 16);
}

#if HAVE_TDM
// One TDM copy of 64 rows x 128 bf16 (16 KB contiguous) from global to LDS,
// padding 4 dwords every 64 dwords => padded row stride 136 ushorts.
static __device__ __forceinline__ void tdm_issue(unsigned lds_off,
                                                 const unsigned short* gsrc) {
    const unsigned long long ga = (unsigned long long)(size_t)gsrc;
    u32x4 g0;
    g0[0] = 1u;                                    // count=1, user desc
    g0[1] = lds_off;                               // lds_addr
    g0[2] = (unsigned)ga;                          // global_addr[31:0]
    g0[3] = (unsigned)((ga >> 32) & 0x01FFFFFFu)   // global_addr[56:32]
          | (2u << 30);                            // type=2 (image)
    i32x8 g1;
    g1[0] = (int)((1u << 16)    // data_size = 2B
                | (1u << 20)    // pad_enable
                | (5u << 22)    // pad_interval: 64 dwords
                | (3u << 25));  // pad_amount:   4 dwords
    g1[1] = (int)(8192u << 16); // tensor_dim0 = 8192 (low16)
    g1[2] = (int)(1u << 16);    // tensor_dim0 hi | tensor_dim1 = 1
    g1[3] = (int)(8192u << 16); // tensor_dim1 hi | tile_dim0 = 8192
    g1[4] = 1;                  // tile_dim1 = 1, tile_dim2 = 0
    g1[5] = 8192;               // tensor_dim0_stride
    g1[6] = 0;
    g1[7] = 0;
    i32x4 gz = {};
#if TDM_6ARG
    i32x8 g4 = {};
    __builtin_amdgcn_tensor_load_to_lds(g0, g1, gz, gz, g4, 0);
#else
    __builtin_amdgcn_tensor_load_to_lds(g0, g1, gz, gz, 0);
#endif
}
#endif

// ---------------------------------------------------------------------------
// K1: q = SCALE * Wqk[:512] @ fmap ; v = Wv @ fmap   (1x1 conv channel-GEMM)
// ---------------------------------------------------------------------------
__global__ __launch_bounds__(256) void qv_kernel(
    const float* __restrict__ fmap,   // [2][128][HW]
    const float* __restrict__ Wqk,    // [1024][128] (rows 0..511 used)
    const float* __restrict__ Wv,     // [512][128]
    unsigned short* __restrict__ Qb,  // [8][HW][128] bf16
    unsigned short* __restrict__ Vb)  // [8][HW][128] bf16
{
    const int d    = threadIdx.x & 127;
    const int isub = threadIdx.x >> 7;
    const int bn   = blockIdx.y;
    const int b    = bn >> 2, n = bn & 3;
    const int i    = blockIdx.x * 2 + isub;
    const int o    = n * 128 + d;
    const float* f = fmap + (size_t)b * 128 * HW + i;
    float q = 0.f, v = 0.f;
#pragma unroll 8
    for (int c = 0; c < 128; ++c) {
        const float fv = f[(size_t)c * HW];
        q += Wqk[(size_t)o * 128 + c] * fv;
        v += Wv [(size_t)o * 128 + c] * fv;
    }
    const size_t idx = ((size_t)bn * HW + i) * 128 + d;
    Qb[idx] = f2bf(q * 0.08838834764831845f);   // 128^-0.5 pre-applied
    Vb[idx] = f2bf(v);
}

// ---------------------------------------------------------------------------
// K2: hs[48]/ws[64] dots + factored softmax -> ph[48], pw[64]
// ---------------------------------------------------------------------------
__global__ __launch_bounds__(128) void probs_kernel(
    const unsigned short* __restrict__ Qb,
    const float* __restrict__ rel_h,        // [199][128]
    const float* __restrict__ rel_w,        // [199][128]
    float* __restrict__ Ph,                 // [8][HW][48]
    float* __restrict__ Pw)                 // [8][HW][64]
{
    __shared__ float qs[128];
    __shared__ float sh[HH], sw[WW];
    __shared__ float red[4];
    const int i   = blockIdx.x;
    const int pix = i % HW;
    const int x   = pix >> 6, y = pix & 63;
    const int t   = threadIdx.x;

    qs[t] = bf2f(Qb[(size_t)i * 128 + t]);
    __syncthreads();

    if (t < HH) {
        const float* r = rel_h + (size_t)(x - t + 99) * 128;
        float s = 0.f;
#pragma unroll 8
        for (int d = 0; d < 128; ++d) s += qs[d] * r[d];
        sh[t] = s;
    } else if (t < HH + WW) {
        const int v = t - HH;
        const float* r = rel_w + (size_t)(y - v + 99) * 128;
        float s = 0.f;
#pragma unroll 8
        for (int d = 0; d < 128; ++d) s += qs[d] * r[d];
        sw[v] = s;
    }
    __syncthreads();
    if (t == 0) {
        float m = sh[0];
        for (int u = 1; u < HH; ++u) m = fmaxf(m, sh[u]);
        float z = 0.f;
        for (int u = 0; u < HH; ++u) z += __expf(sh[u] - m);
        red[0] = m; red[1] = z;
    } else if (t == 1) {
        float m = sw[0];
        for (int v = 1; v < WW; ++v) m = fmaxf(m, sw[v]);
        float z = 0.f;
        for (int v = 0; v < WW; ++v) z += __expf(sw[v] - m);
        red[2] = m; red[3] = z;
    }
    __syncthreads();
    if (t < HH)
        Ph[(size_t)i * HH + t] = __expf(sh[t] - red[0]) / red[1];
    else if (t < HH + WW)
        Pw[(size_t)i * WW + (t - HH)] = __expf(sw[t - HH] - red[2]) / red[3];
}

// ---------------------------------------------------------------------------
// K3 (WMMA core): O[bn][q][d] = sum_u ph[q][u] * ( sum_v pw[q][v] V[u*64+v][d] )
// A fragments (pure pw, bf16) are loop-invariant and STATICALLY indexed so
// they live in VGPRs (no scratch spill). Per key-row u: two chained WMMAs
// (C-operand accumulation across the two v-halves), then one ph-scaled FMA.
// V staged 64 rows/iter via TDM (async, double-buffered, HW-padded stride).
// ---------------------------------------------------------------------------
__global__ __launch_bounds__(256, 1) void attn_av_kernel(
    const unsigned short* __restrict__ Vb,  // [8][HW][128] bf16
    const float* __restrict__ Ph,           // [8][HW][48]
    const float* __restrict__ Pw,           // [8][HW][64]
    float* __restrict__ O)                  // [8][HW][128]
{
    __shared__ unsigned short Vs[2][64][136];  // 2 x 64 padded rows (17 KB ea)
    __shared__ float          phs[64][HH];
    __shared__ unsigned short pwb[64][WW];     // pw as bf16

    const int bn  = blockIdx.y;
    const int q0  = blockIdx.x * 64;
    const int tid = threadIdx.x;

    for (int t = tid; t < 64 * HH; t += 256) {
        const int m = t / HH, u = t - m * HH;
        phs[m][u] = Ph[((size_t)bn * HW + q0 + m) * HH + u];
    }
    for (int t = tid; t < 64 * WW; t += 256) {
        const int m = t >> 6, v = t & 63;
        pwb[m][v] = f2bf(Pw[((size_t)bn * HW + q0 + m) * WW + v]);
    }
    __syncthreads();

    const int lane = tid & 31;
    const int wid  = tid >> 5;
    const int n0   = wid * 16;
    const int grp  = lane >> 4;
    const int ln   = lane & 15;

    // Loop-invariant A fragments, always statically indexed below.
    v16bf afrag0[4], afrag1[4];      // [m-subtile] for v-half 0 / 1
#pragma unroll
    for (int ms = 0; ms < 4; ++ms) {
        const int m = ms * 16 + ln;
        v16u a0, a1;
#pragma unroll
        for (int e = 0; e < 16; ++e) {
            const int kl = ((e >= 8) ? 16 : 0) + grp * 8
                         + (((e >> 1) & 3) * 2) + (e & 1);
            a0[e] = pwb[m][kl];
            a1[e] = pwb[m][32 + kl];
        }
        afrag0[ms] = __builtin_bit_cast(v16bf, a0);
        afrag1[ms] = __builtin_bit_cast(v16bf, a1);
    }

    v8f acc[4] = {};
    const v8f zeroC = {};
    const unsigned short* Vbn = Vb + (size_t)bn * HW * 128;

#if HAVE_TDM
    const unsigned vsoff0 = (unsigned)(size_t)&Vs[0][0][0];
    const unsigned vsoff1 = (unsigned)(size_t)&Vs[1][0][0];
    if (wid == 0) tdm_issue(vsoff0, Vbn);          // prologue: row 0 -> buf 0
#endif

    for (int u = 0; u < HH; ++u) {
        __syncthreads();            // all waves done reading buf[(u-1)&1]
#if HAVE_TDM
        if (wid == 0) {
            if (u + 1 < HH) {
                tdm_issue(((u + 1) & 1) ? vsoff1 : vsoff0,
                          Vbn + (size_t)(u + 1) * 64 * 128);
                __builtin_amdgcn_s_wait_tensorcnt(1);   // row u landed
            } else {
                __builtin_amdgcn_s_wait_tensorcnt(0);
            }
        }
#else
        {
            const uint4* src = (const uint4*)(Vbn + (size_t)u * 64 * 128);
            uint4* dst = (uint4*)&Vs[u & 1][0][0];
            // 64 rows x 128 bf16 = 1024 uint4 chunks; padded dest rows
#pragma unroll
            for (int c = tid; c < 1024; c += 256) {
                const int row = c >> 4;
                const int col = (c & 15) * 8;
                *(uint4*)&Vs[u & 1][row][col] = src[c];
            }
            (void)dst;
            if (u + 1 < HH)
                __builtin_prefetch(Vbn + (size_t)(u + 1) * 64 * 128, 0, 1);
        }
#endif
        __syncthreads();

        const unsigned short* bufp = &Vs[u & 1][0][0];
        // B fragments for v-halves 0/1: element e -> K = grp*8+e (+8 if e>=8), N = n0+ln
        v16u bu0, bu1;
#pragma unroll
        for (int e = 0; e < 16; ++e) {
            bu0[e] = bufp[(size_t)(grp * 8 + e + ((e >= 8) ? 8 : 0)) * 136 + n0 + ln];
            bu1[e] = bufp[(size_t)(32 + grp * 8 + e + ((e >= 8) ? 8 : 0)) * 136 + n0 + ln];
        }
        const v16bf bfrag0 = __builtin_bit_cast(v16bf, bu0);
        const v16bf bfrag1 = __builtin_bit_cast(v16bf, bu1);

#pragma unroll
        for (int ms = 0; ms < 4; ++ms) {
            // E[m][d] = sum_{v} pw[m][v] * V[u*64+v][d]  (chained through C)
            v8f E = __builtin_amdgcn_wmma_f32_16x16x32_bf16(
                false, afrag0[ms], false, bfrag0, (short)0, zeroC,
                false, false);
            E = __builtin_amdgcn_wmma_f32_16x16x32_bf16(
                false, afrag1[ms], false, bfrag1, (short)0, E,
                false, false);
            const float* prow = &phs[ms * 16 + grp * 8][u];
#pragma unroll
            for (int r = 0; r < 8; ++r)
                acc[ms][r] += prow[(size_t)r * HH] * E[r];
        }
    }

    // epilogue: C/D layout -> row m = r + grp*8, col n = ln
    float* Obn = O + (size_t)bn * HW * 128;
#pragma unroll
    for (int ms = 0; ms < 4; ++ms) {
#pragma unroll
        for (int r = 0; r < 8; ++r) {
            const int m = q0 + ms * 16 + grp * 8 + r;
            Obn[(size_t)m * 128 + n0 + ln] = acc[ms][r];
        }
    }
}

// ---------------------------------------------------------------------------
// K4: out = fmap + gamma * (Wp @ O_channels)  ; O staged through LDS
// ---------------------------------------------------------------------------
__global__ __launch_bounds__(256) void proj_kernel(
    const float* __restrict__ fmap,   // [2][128][HW]
    const float* __restrict__ Wp,     // [128][512]
    const float* __restrict__ Og,     // [8][HW][128]
    const float* __restrict__ gamma,  // [1]
    float* __restrict__ out)          // [2][128][HW]
{
    __shared__ float Os[2][512];
    const int b   = blockIdx.y;
    const int i0  = blockIdx.x * 2;
    const int tid = threadIdx.x;
    for (int t = tid; t < 1024; t += 256) {
        const int p  = t >> 9;
        const int o  = t & 511;
        const int nn = o >> 7, d = o & 127;
        Os[p][o] = Og[(((size_t)(b * 4 + nn)) * HW + (i0 + p)) * 128 + d];
    }
    __syncthreads();
    const int p  = tid >> 7;
    const int ch = tid & 127;
    const float* Wrow = Wp + (size_t)ch * 512;
    float s = 0.f;
#pragma unroll 8
    for (int o = 0; o < 512; ++o) s += Wrow[o] * Os[p][o];
    const size_t oi = ((size_t)b * 128 + ch) * HW + i0 + p;
    out[oi] = fmap[oi] + gamma[0] * s;
}

// ---------------------------------------------------------------------------
extern "C" void kernel_launch(void* const* d_in, const int* in_sizes, int n_in,
                              void* d_out, int out_size, void* d_ws, size_t ws_size,
                              hipStream_t stream) {
    (void)in_sizes; (void)n_in; (void)out_size; (void)ws_size;
    const float* fmaps[2] = { (const float*)d_in[0], (const float*)d_in[1] };
    const float* Wqk   = (const float*)d_in[2];
    const float* Wv    = (const float*)d_in[3];
    const float* rel_h = (const float*)d_in[4];
    const float* rel_w = (const float*)d_in[5];
    const float* Wp    = (const float*)d_in[6];
    const float* gamma = (const float*)d_in[7];

    // workspace carve-up (reused across the two maps): ~34.5 MB total
    char* ws = (char*)d_ws;
    unsigned short* Qb = (unsigned short*)ws; ws += (size_t)NBN * HW * 128 * 2;
    unsigned short* Vb = (unsigned short*)ws; ws += (size_t)NBN * HW * 128 * 2;
    float* Ph = (float*)ws;                   ws += (size_t)NBN * HW * HH  * 4;
    float* Pw = (float*)ws;                   ws += (size_t)NBN * HW * WW  * 4;
    float* Og = (float*)ws;                   ws += (size_t)NBN * HW * 128 * 4;

    for (int map = 0; map < 2; ++map) {
        float* out = (float*)d_out + (size_t)map * 2 * 128 * HW;
        qv_kernel     <<<dim3(HW / 2, NBN),  256, 0, stream>>>(fmaps[map], Wqk, Wv, Qb, Vb);
        probs_kernel  <<<dim3(NBN * HW),     128, 0, stream>>>(Qb, rel_h, rel_w, Ph, Pw);
        attn_av_kernel<<<dim3(HW / 64, NBN), 256, 0, stream>>>(Vb, Ph, Pw, Og);
        proj_kernel   <<<dim3(HW / 2, 2),    256, 0, stream>>>(fmaps[map], Wp, Og, gamma, out);
    }
}
